// SwinTransformerBlock_AdaFM_25537875542336
// MI455X (gfx1250) — matrix operand order repaired and checked by reading, not compiled, not run on hardware
//
#include <hip/hip_runtime.h>
#include <hip/hip_bf16.h>

typedef __attribute__((ext_vector_type(16))) _Float16 v16h;
typedef __attribute__((ext_vector_type(8)))  float    v8f;
typedef _Float16 h16;

#define TID ((int)threadIdx.x)

// ---------------------------------------------------------------------------
// CDNA5 async global->LDS copy (guarded; falls back to sync copy)
// ---------------------------------------------------------------------------
#if defined(__HIP_DEVICE_COMPILE__) && \
    __has_builtin(__builtin_amdgcn_global_load_async_to_lds_b128) && \
    __has_builtin(__builtin_amdgcn_s_wait_asynccnt)
#define USE_ASYNC 1
#else
#define USE_ASYNC 0
#endif

#if USE_ASYNC
typedef int i4 __attribute__((vector_size(16)));
typedef __attribute__((address_space(1))) i4* as1_i4p;
typedef __attribute__((address_space(3))) i4* as3_i4p;
#endif

__device__ __forceinline__ void cp16(const h16* g, h16* l) {
#if USE_ASYNC
  __builtin_amdgcn_global_load_async_to_lds_b128(
      (as1_i4p)(i4*)const_cast<h16*>(g),
      (as3_i4p)(i4*)l, 0, 0);
#else
  *reinterpret_cast<uint4*>(l) = *reinterpret_cast<const uint4*>(g);
#endif
}

#if USE_ASYNC
#define ASYNC_WAIT(n) __builtin_amdgcn_s_wait_asynccnt(n)
#else
#define ASYNC_WAIT(n) ((void)0)
#endif

__constant__ float CTAB[8] = {1.0f, 0.70710678f, 0.0f, -0.70710678f,
                              -1.0f, -0.70710678f, 0.0f, 0.70710678f};

static constexpr float GN_EPS = 1e-5f;
static constexpr float SCALEQ = 0.17677669529663687f;  // (192/6)^-0.5

// ---------------------------------------------------------------------------
// WMMA fragment loaders (CDNA5 16-bit A 16x32 / B 32x16 per-lane layouts)
// ---------------------------------------------------------------------------
__device__ __forceinline__ v16h load_afrag(const h16* s, int row0, int stride, int k0) {
  const int lane = TID & 31, hl = lane >> 4, r = lane & 15;
  const h16* row = s + (row0 + r) * stride;
  v16h a;
#pragma unroll
  for (int j = 0; j < 8; ++j) {
    const int k = k0 + ((j < 4) ? (2 * j) : (16 + 2 * (j - 4))) + 8 * hl;
    a[2 * j] = row[k];
    a[2 * j + 1] = row[k + 1];
  }
  return a;
}

__device__ __forceinline__ v16h load_bfrag(const h16* s, int n0, int stride, int k0) {
  const int lane = TID & 31, hl = lane >> 4, r = lane & 15;
  const h16* row = s + (n0 + r) * stride;
  v16h b;
#pragma unroll
  for (int j = 0; j < 8; ++j) {
    const int k = k0 + ((j < 4) ? (2 * j) : (16 + 2 * (j - 4))) + 8 * hl;
    b[2 * j] = row[k];
    b[2 * j + 1] = row[k + 1];
  }
  return b;
}

// ---------------------------------------------------------------------------
// silu(t) -> s (40 freq) -> full 8x8 spectrum -> circular AdaFM kernel Kc[8][8]
// ---------------------------------------------------------------------------
__global__ __launch_bounds__(64) void silu_ada(const float* __restrict__ t,
    const float* __restrict__ wmsa, const float* __restrict__ bmsa,
    const float* __restrict__ wmlp, const float* __restrict__ bmlp,
    float* __restrict__ KcOut) {
  __shared__ float sh[40];
  __shared__ float sf[64];
  const int b = blockIdx.x >> 1, br = blockIdx.x & 1;
  const float* wp = br ? wmlp : wmsa;
  const float* bp = br ? bmlp : bmsa;
  if (TID < 40) {
    float acc = bp[TID];
    const float* wr = wp + TID * 640;
    const float* tr = t + b * 640;
    for (int e = 0; e < 640; ++e) {
      const float tv = tr[e];
      acc += (tv / (1.0f + __expf(-tv))) * wr[e];
    }
    sh[TID] = acc;
  }
  __syncthreads();
  {
    const int u = TID >> 3, v = TID & 7;
    sf[TID] = (v <= 4) ? sh[u * 5 + v] : sh[((8 - u) & 7) * 5 + (8 - v)];
  }
  __syncthreads();
  const int dr = TID >> 3, dc = TID & 7;
  float y = 0.0f;
#pragma unroll
  for (int u = 0; u < 8; ++u)
#pragma unroll
    for (int v = 0; v < 8; ++v)
      y += sf[u * 8 + v] * CTAB[(u * dr + v * dc) & 7];
  KcOut[((size_t)br * 32 + b) * 64 + TID] = y * (1.0f / 64.0f);
}

// ---------------------------------------------------------------------------
// GroupNorm statistics: one block per (b, group)
// ---------------------------------------------------------------------------
__global__ __launch_bounds__(256) void gn_stats(const float* __restrict__ src,
                                                float* __restrict__ stats) {
  __shared__ float r1[256], r2[256];
  const int b = blockIdx.x >> 5, g = blockIdx.x & 31;
  const float* p = src + ((size_t)b * 192 + g * 6) * 4096;
  float s = 0.0f, s2 = 0.0f;
  for (int i = TID; i < 24576; i += 256) {
    const float v = p[i];
    s += v; s2 += v * v;
  }
  r1[TID] = s; r2[TID] = s2;
  __syncthreads();
  for (int st = 128; st > 0; st >>= 1) {
    if (TID < st) { r1[TID] += r1[TID + st]; r2[TID] += r2[TID + st]; }
    __syncthreads();
  }
  if (TID == 0) {
    const float mu = r1[0] * (1.0f / 24576.0f);
    const float var = r2[0] * (1.0f / 24576.0f) - mu * mu;
    stats[blockIdx.x * 2] = mu;
    stats[blockIdx.x * 2 + 1] = rsqrtf(var + GN_EPS);
  }
}

// ---------------------------------------------------------------------------
// GroupNorm-apply + AdaFM circular conv + layout write (f16).
// mode 0: roll(-4,-4) + window-partition token-major; mode 1: token-major
// ---------------------------------------------------------------------------
__global__ __launch_bounds__(256) void norm_adafm(const float* __restrict__ src,
    const float* __restrict__ stats, const float* __restrict__ gamma,
    const float* __restrict__ beta, const float* __restrict__ Kc,
    h16* __restrict__ dst, int mode) {
  __shared__ float pl[4096];
  __shared__ float kc[64];
  const int b = blockIdx.x / 192, c = blockIdx.x % 192;
  const int g = c / 6;
  const float mu = stats[(b * 32 + g) * 2];
  const float rs = stats[(b * 32 + g) * 2 + 1];
  const float ga = gamma[c], be = beta[c];
  const float* sp = src + ((size_t)b * 192 + c) * 4096;
  for (int i = TID; i < 4096; i += 256) pl[i] = (sp[i] - mu) * rs * ga + be;
  if (TID < 64) kc[TID] = Kc[b * 64 + TID];
  __syncthreads();
  for (int i = TID; i < 4096; i += 256) {
    const int hh = i >> 6, wwp = i & 63, pr = hh & 7, pc = wwp & 7;
    const int base = (hh & 0x38) * 64 + (wwp & 0x38);
    float y = 0.0f;
#pragma unroll
    for (int qr = 0; qr < 8; ++qr) {
      const float* kr = &kc[((pr - qr) & 7) * 8];
      const float* xr = &pl[base + qr * 64];
#pragma unroll
      for (int qc = 0; qc < 8; ++qc) y += kr[(pc - qc) & 7] * xr[qc];
    }
    size_t o;
    if (mode == 0) {
      const int h2 = (hh - 4) & 63, w2 = (wwp - 4) & 63;
      const int widx = ((h2 >> 3) << 3) + (w2 >> 3);
      const int tok = ((h2 & 7) << 3) + (w2 & 7);
      o = ((size_t)b * 4096 + widx * 64 + tok) * 192 + c;
    } else {
      o = ((size_t)b * 4096 + hh * 64 + wwp) * 192 + c;
    }
    dst[o] = (h16)y;
  }
}

// ---------------------------------------------------------------------------
// Weight prep: src (Nn,Kk) f32 row-major weight; logical B[k][n] = src[n*Kk+k]
// Output WMMA-ready tiles: dst[((np*(Kk/32) + kp)*64 + nl)*32 + kl]
// ---------------------------------------------------------------------------
__global__ void prep_b_tiled(const float* __restrict__ src, h16* __restrict__ dst,
                             int Nn, int Kk) {
  const int idx = blockIdx.x * 256 + TID;
  if (idx < Nn * Kk) {
    const int n = idx / Kk, k = idx % Kk;
    const int np = n >> 6, nl = n & 63, kp = k >> 5, kl = k & 31;
    const int nkt = Kk >> 5;
    dst[(((size_t)np * nkt + kp) * 64 + nl) * 32 + kl] = (h16)src[idx];
  }
}

// ---------------------------------------------------------------------------
// Generic WMMA GEMM, async double-buffered staging.
// Block tile 128x64, 8 waves, 32x32 per wave, K-step 32.
// ---------------------------------------------------------------------------
enum { EPI_QKV = 0, EPI_PROJ = 1, EPI_GELU = 2, EPI_FC2 = 3 };

__device__ __forceinline__ void stage_tile(const h16* A, const h16* Bt, int K,
                                           int m0, int ks, int ar, int akc,
                                           int bnl, int bkp,
                                           h16* Asb, h16* Bsb) {
  const h16* ga = A + (size_t)(m0 + ar) * K + ks * 32 + akc;
  h16* la = Asb + ar * 40 + akc;
  cp16(ga, la);
  cp16(ga + 8, la + 8);
  const h16* gb = Bt + (size_t)ks * 2048 + bnl * 32 + bkp;
  cp16(gb, Bsb + bnl * 40 + bkp);
}

template <int EPI>
__global__ __launch_bounds__(256) void gemm_wmma(
    const h16* __restrict__ A, const h16* __restrict__ Bw,
    const float* __restrict__ bias, int K, int N,
    h16* __restrict__ outH, float* __restrict__ outF,
    const float* __restrict__ addSrc) {
  __shared__ __align__(16) h16 As[2][128 * 40];
  __shared__ __align__(16) h16 BsT[2][64 * 40];
  const int m0 = blockIdx.x * 128;
  const int n0 = blockIdx.y * 64;
  const int tid = TID;
  const int ar = tid >> 1, akc = (tid & 1) * 16;      // A: 16 halves / thread
  const int bnl = tid >> 2, bkp = (tid & 3) * 8;      // B: 8 halves / thread
  const int wv = tid >> 5, wm = (wv >> 1) * 32, wn = (wv & 1) * 32;
  const int lane = tid & 31, hl = lane >> 4, lr = lane & 15;
  const int nk = K / 32;
  const h16* Bt = Bw + (size_t)(n0 >> 6) * (size_t)nk * 2048;
  v8f acc[2][2] = {};

  stage_tile(A, Bt, K, m0, 0, ar, akc, bnl, bkp, As[0], BsT[0]);
  for (int ks = 0; ks < nk; ++ks) {
    const int cur = ks & 1;
    if (ks + 1 < nk) {
      stage_tile(A, Bt, K, m0, ks + 1, ar, akc, bnl, bkp, As[cur ^ 1], BsT[cur ^ 1]);
      ASYNC_WAIT(3);   // oldest 3-op group (tile ks) complete
    } else {
      ASYNC_WAIT(0);
    }
    __syncthreads();
    const v16h a0 = load_afrag(As[cur], wm, 40, 0);
    const v16h a1 = load_afrag(As[cur], wm + 16, 40, 0);
    const v16h b0 = load_bfrag(BsT[cur], wn, 40, 0);
    const v16h b1 = load_bfrag(BsT[cur], wn + 16, 40, 0);
    acc[0][0] = __builtin_amdgcn_wmma_f32_16x16x32_f16(false, a0, false, b0, (short)0, acc[0][0], false, false);
    acc[0][1] = __builtin_amdgcn_wmma_f32_16x16x32_f16(false, a0, false, b1, (short)0, acc[0][1], false, false);
    acc[1][0] = __builtin_amdgcn_wmma_f32_16x16x32_f16(false, a1, false, b0, (short)0, acc[1][0], false, false);
    acc[1][1] = __builtin_amdgcn_wmma_f32_16x16x32_f16(false, a1, false, b1, (short)0, acc[1][1], false, false);
    __syncthreads();
  }
#pragma unroll
  for (int mi = 0; mi < 2; ++mi)
#pragma unroll
    for (int ni = 0; ni < 2; ++ni)
#pragma unroll
      for (int rr = 0; rr < 8; ++rr) {
        const int m = m0 + wm + mi * 16 + rr + 8 * hl;
        const int n = n0 + wn + ni * 16 + lr;
        float y = acc[mi][ni][rr] + bias[n];
        if constexpr (EPI == EPI_QKV) {
          if (n < 192) y *= SCALEQ;  // scale q (incl. bias, as in reference)
          outH[(size_t)m * 576 + n] = (h16)y;
        } else if constexpr (EPI == EPI_PROJ) {
          // window reverse + roll(+4,+4) scatter with residual
          const int b = m >> 12, rem = m & 4095, wid = rem >> 6, tok = rem & 63;
          const int hh = ((((wid >> 3) << 3) + (tok >> 3)) + 4) & 63;
          const int wwp = ((((wid & 7) << 3) + (tok & 7)) + 4) & 63;
          const size_t idx = (((size_t)b * 192 + n) * 64 + hh) * 64 + wwp;
          outF[idx] = addSrc[idx] + y;
        } else if constexpr (EPI == EPI_GELU) {
          const float gv = 0.5f * y * (1.0f + erff(y * 0.70710678f));
          outH[(size_t)m * 768 + n] = (h16)gv;
        } else {  // EPI_FC2: accumulate into residual (d_out)
          const int b = m >> 12, pos = m & 4095;
          const size_t idx = ((size_t)b * 192 + n) * 4096 + pos;
          outF[idx] += y;
        }
      }
}

// ---------------------------------------------------------------------------
// Windowed attention: one block (128 thr = 4 waves) per (window, head).
// S = q*k^T (+rpb, +shift-mask), softmax, out = P*v. All WMMA in LDS.
// ---------------------------------------------------------------------------
__global__ __launch_bounds__(128) void attn_kernel(const h16* __restrict__ qkv,
                                                   const float* __restrict__ rpb,
                                                   h16* __restrict__ ctx) {
  __shared__ __align__(16) h16 qs[64 * 40];
  __shared__ __align__(16) h16 ksh[64 * 40];
  __shared__ __align__(16) h16 vsT[32 * 72];
  __shared__ float Ss[64 * 72];
  __shared__ __align__(16) h16 Ps[64 * 72];
  const int win = blockIdx.x, head = blockIdx.y, wid = win & 63;
  const int tid = TID;
  {
    const int tok = tid >> 1, d0 = (tid & 1) * 16;
    const h16* base = qkv + ((size_t)win * 64 + tok) * 576 + head * 32 + d0;
    h16* qdst = &qs[tok * 40 + d0];
    h16* kdst = &ksh[tok * 40 + d0];
    cp16(base, qdst);        cp16(base + 8, qdst + 8);       // q (async)
    cp16(base + 192, kdst);  cp16(base + 200, kdst + 8);     // k (async)
#pragma unroll
    for (int j = 0; j < 16; ++j) vsT[(d0 + j) * 72 + tok] = base[384 + j];
  }
  ASYNC_WAIT(0);
  __syncthreads();
  const int wv = tid >> 5, lane = tid & 31, hl = lane >> 4, lr = lane & 15;
  const int qrow0 = wv * 16;
  const v16h aq = load_afrag(qs, qrow0, 40, 0);
#pragma unroll
  for (int nt = 0; nt < 4; ++nt) {
    const v16h bk = load_bfrag(ksh, nt * 16, 40, 0);
    v8f s = {};
    s = __builtin_amdgcn_wmma_f32_16x16x32_f16(false, aq, false, bk, (short)0, s, false, false);
#pragma unroll
    for (int rr = 0; rr < 8; ++rr) {
      const int qt = qrow0 + rr + 8 * hl;
      const int kt = nt * 16 + lr;
      const int qi = qt >> 3, qj = qt & 7, ki = kt >> 3, kj = kt & 7;
      const float bv = rpb[((qi - ki + 7) * 15 + (qj - kj + 7)) * 6 + head];
      const int qh = ((wid >> 3) << 3) + qi, qw = ((wid & 7) << 3) + qj;
      const int kh = ((wid >> 3) << 3) + ki, kw = ((wid & 7) << 3) + kj;
      const int lq = (qh < 56 ? 0 : (qh < 60 ? 1 : 2)) * 3 + (qw < 56 ? 0 : (qw < 60 ? 1 : 2));
      const int lk = (kh < 56 ? 0 : (kh < 60 ? 1 : 2)) * 3 + (kw < 56 ? 0 : (kw < 60 ? 1 : 2));
      const float mk = (lq != lk) ? -100.0f : 0.0f;
      Ss[qt * 72 + kt] = s[rr] + bv + mk;
    }
  }
  __syncthreads();
  if (tid < 64) {  // row softmax
    float mx = -1e30f;
    for (int j = 0; j < 64; ++j) mx = fmaxf(mx, Ss[tid * 72 + j]);
    float sum = 0.0f;
    for (int j = 0; j < 64; ++j) {
      const float e = __expf(Ss[tid * 72 + j] - mx);
      Ss[tid * 72 + j] = e;
      sum += e;
    }
    const float inv = 1.0f / sum;
    for (int j = 0; j < 64; ++j) Ps[tid * 72 + j] = (h16)(Ss[tid * 72 + j] * inv);
  }
  __syncthreads();
#pragma unroll
  for (int dt = 0; dt < 2; ++dt) {
    v8f o = {};
#pragma unroll
    for (int kk = 0; kk < 2; ++kk) {
      const v16h ap = load_afrag(Ps, qrow0, 72, kk * 32);
      const v16h bv = load_bfrag(vsT, dt * 16, 72, kk * 32);
      o = __builtin_amdgcn_wmma_f32_16x16x32_f16(false, ap, false, bv, (short)0, o, false, false);
    }
#pragma unroll
    for (int rr = 0; rr < 8; ++rr) {
      const int t = qrow0 + rr + 8 * hl;
      const int d = dt * 16 + lr;
      ctx[((size_t)win * 64 + t) * 192 + head * 32 + d] = (h16)o[rr];
    }
  }
}

// ---------------------------------------------------------------------------
extern "C" void kernel_launch(void* const* d_in, const int* in_sizes, int n_in,
                              void* d_out, int out_size, void* d_ws, size_t ws_size,
                              hipStream_t stream) {
  const float* x      = (const float*)d_in[0];
  const float* t      = (const float*)d_in[1];
  const float* n1_w   = (const float*)d_in[2];
  const float* n1_b   = (const float*)d_in[3];
  const float* qkv_w  = (const float*)d_in[4];
  const float* qkv_b  = (const float*)d_in[5];
  const float* rpb    = (const float*)d_in[6];
  const float* proj_w = (const float*)d_in[7];
  const float* proj_b = (const float*)d_in[8];
  const float* n2_w   = (const float*)d_in[9];
  const float* n2_b   = (const float*)d_in[10];
  const float* fc1_w  = (const float*)d_in[11];
  const float* fc1_b  = (const float*)d_in[12];
  const float* fc2_w  = (const float*)d_in[13];
  const float* fc2_b  = (const float*)d_in[14];
  const float* amw    = (const float*)d_in[15];
  const float* amb    = (const float*)d_in[16];
  const float* apw    = (const float*)d_in[17];
  const float* apb    = (const float*)d_in[18];
  float* out = (float*)d_out;
  char* ws = (char*)d_ws;

  const size_t OFF_KC   = 0;                                   // 2*32*64 f32
  const size_t OFF_ST1  = 16 * 1024;                           // 1024*2 f32
  const size_t OFF_ST2  = 24 * 1024;
  const size_t OFF_WQKV = 64 * 1024;                           // 576*192 f16 tiled
  const size_t OFF_WPROJ = OFF_WQKV + (size_t)576 * 192 * 2;
  const size_t OFF_WFC1  = OFF_WPROJ + (size_t)192 * 192 * 2;
  const size_t OFF_WFC2  = OFF_WFC1 + (size_t)768 * 192 * 2;
  const size_t OFF_A = (size_t)1 << 20;                        // hwin / h2t f16
  const size_t OFF_B = OFF_A + (size_t)131072 * 192 * 2;       // qkv f16, later fc1out
  const size_t OFF_C = OFF_B + (size_t)131072 * 576 * 2;       // ctx f16
  const size_t NEED  = OFF_B + (size_t)131072 * 768 * 2;       // ~241 MiB
  if (ws_size < NEED) return;

  float* Kc    = (float*)(ws + OFF_KC);
  float* st1   = (float*)(ws + OFF_ST1);
  float* st2   = (float*)(ws + OFF_ST2);
  h16* wqkvT   = (h16*)(ws + OFF_WQKV);
  h16* wprojT  = (h16*)(ws + OFF_WPROJ);
  h16* wfc1T   = (h16*)(ws + OFF_WFC1);
  h16* wfc2T   = (h16*)(ws + OFF_WFC2);
  h16* hwin    = (h16*)(ws + OFF_A);   // also h2t
  h16* qkvbuf  = (h16*)(ws + OFF_B);
  h16* ctx     = (h16*)(ws + OFF_C);
  h16* fc1out  = (h16*)(ws + OFF_B);   // overlaps dead qkv+ctx

  silu_ada<<<64, 64, 0, stream>>>(t, amw, amb, apw, apb, Kc);
  prep_b_tiled<<<(576 * 192 + 255) / 256, 256, 0, stream>>>(qkv_w, wqkvT, 576, 192);
  prep_b_tiled<<<(192 * 192 + 255) / 256, 256, 0, stream>>>(proj_w, wprojT, 192, 192);
  prep_b_tiled<<<(768 * 192 + 255) / 256, 256, 0, stream>>>(fc1_w, wfc1T, 768, 192);
  prep_b_tiled<<<(192 * 768 + 255) / 256, 256, 0, stream>>>(fc2_w, wfc2T, 192, 768);

  gn_stats<<<1024, 256, 0, stream>>>(x, st1);
  norm_adafm<<<6144, 256, 0, stream>>>(x, st1, n1_w, n1_b, Kc, hwin, 0);
  gemm_wmma<EPI_QKV><<<dim3(1024, 9), 256, 0, stream>>>(hwin, wqkvT, qkv_b, 192, 576, qkvbuf, nullptr, nullptr);
  attn_kernel<<<dim3(2048, 6), 128, 0, stream>>>(qkvbuf, rpb, ctx);
  gemm_wmma<EPI_PROJ><<<dim3(1024, 3), 256, 0, stream>>>(ctx, wprojT, proj_b, 192, 192, nullptr, out, x);

  gn_stats<<<1024, 256, 0, stream>>>(out, st2);
  norm_adafm<<<6144, 256, 0, stream>>>(out, st2, n2_w, n2_b, Kc + 32 * 64, hwin, 1);
  gemm_wmma<EPI_GELU><<<dim3(1024, 12), 256, 0, stream>>>(hwin, wfc1T, fc1_b, 192, 768, fc1out, nullptr, nullptr);
  gemm_wmma<EPI_FC2><<<dim3(1024, 3), 256, 0, stream>>>(fc1out, wfc2T, fc2_b, 768, 192, nullptr, out, nullptr);

  (void)in_sizes; (void)n_in; (void)out_size;
}
